// InitModulatedConv2d_67748814127433
// MI455X (gfx1250) — matrix operand order repaired and checked by reading, not compiled, not run on hardware
//
#include <hip/hip_runtime.h>
#include <hip/hip_bf16.h>

typedef __attribute__((ext_vector_type(16))) __bf16 v16bf;
typedef __attribute__((ext_vector_type(8)))  float  v8f;

#define CIN   512
#define COUT  512
#define NB    16
#define HH    64
#define WW    64
#define KTOT  (CIN * 9)          // 4608
#define NKBLK (9 * (CIN / 32))   // 144 K-blocks of 32
#define W_SCALE   0.014731391274719739f   // 1/sqrt(512*9)
#define LIN_SCALE 0.044194173824159216f   // 1/sqrt(512)

__device__ __forceinline__ unsigned short f2bf(float f) {
    unsigned int u = __float_as_uint(f);
    u += 0x7fffu + ((u >> 16) & 1u);      // round-to-nearest-even
    return (unsigned short)(u >> 16);
}

__device__ __forceinline__ v8f wmma_bf16(v16bf a, v16bf b, v8f c) {
    return __builtin_amdgcn_wmma_f32_16x16x32_bf16(false, a, false, b,
                                                   (short)0, c, false, false);
}

// ---------------------------------------------------------------------------
// Kernel 0: stream-convert input fp32 -> bf16 (makes conv working set fit L2)
// ---------------------------------------------------------------------------
__global__ void cvt_kernel(const float* __restrict__ in,
                           unsigned short* __restrict__ o) {
    int gid = blockIdx.x * blockDim.x + threadIdx.x;   // over n/4 float4s
    const float4 v = ((const float4*)in)[gid];
    ushort4 r;
    r.x = f2bf(v.x); r.y = f2bf(v.y); r.z = f2bf(v.z); r.w = f2bf(v.w);
    ((ushort4*)o)[gid] = r;
}

// ---------------------------------------------------------------------------
// Kernel 1: s_masked[b][c] = mask(b,c) * (style[b]·mod_weight[c]*LIN + bias[c])
// ---------------------------------------------------------------------------
__global__ void style_kernel(const float* __restrict__ style,
                             const float* __restrict__ prob,
                             const float* __restrict__ mw,
                             const float* __restrict__ mb,
                             float* __restrict__ s_out) {
    int gid = blockIdx.x * blockDim.x + threadIdx.x;   // 8192 = 16*512
    int b = gid >> 9, c = gid & 511;
    const float* st = style + b * 512;
    const float* w  = mw + (size_t)c * 512;
    float acc = 0.f;
    for (int k = 0; k < 512; ++k) acc += st[k] * w[k];
    float s = acc * LIN_SCALE + mb[c];
    float thresh = rintf(prob[b] * 8.0f) * 32.0f;      // floor(round(p*8)/8*256)
    s_out[gid] = ((float)c >= thresh) ? s : 0.f;
}

// ---------------------------------------------------------------------------
// Kernel 2: modulate + demod weights; emit bf16 in WMMA A-fragment lane order.
// Fragment id = ((b*144 + kpos*16 + c/32)*32 + cout/16); 512 bf16 per fragment
// laid out so a wave loads it with one contiguous v16bf per lane.
// ---------------------------------------------------------------------------
__global__ void wprep_kernel(const float* __restrict__ weight,
                             const float* __restrict__ s_in,
                             unsigned short* __restrict__ wm) {
    __shared__ float sm[512];
    __shared__ float red[256];
    int o = blockIdx.x, b = blockIdx.y, tid = threadIdx.x;
    sm[tid]       = s_in[b * 512 + tid];
    sm[tid + 256] = s_in[b * 512 + tid + 256];
    __syncthreads();
    const float* wrow = weight + (size_t)o * KTOT;
    float sum = 0.f;
    for (int j = tid; j < KTOT; j += 256) {
        int c = j / 9;
        float v = W_SCALE * wrow[j] * sm[c];
        sum += v * v;
    }
    red[tid] = sum; __syncthreads();
    for (int s = 128; s > 0; s >>= 1) {
        if (tid < s) red[tid] += red[tid + s];
        __syncthreads();
    }
    float demod = rsqrtf(red[0] + 1e-8f);
    for (int j = tid; j < KTOT; j += 256) {
        int c = j / 9, kpos = j - c * 9;
        float v = W_SCALE * wrow[j] * sm[c] * demod;
        // A-fragment map (16-bit A 16x32): K = i + laneHi*16
        int kin    = c & 31;
        int laneHi = (kin >> 4) & 1;
        int i      = kin & 15;
        int lane   = (o & 15) + (laneHi << 4);
        int kblk   = kpos * 16 + (c >> 5);
        size_t adr = ((size_t)((b * NKBLK + kblk) * 32 + (o >> 4))) * 512
                     + lane * 16 + i;
        wm[adr] = f2bf(v);
    }
}

// ---------------------------------------------------------------------------
// Kernel 3: implicit-GEMM conv. Block (512 thr, 16 waves) = (batch b,
// row-pair, 128 couts x 64 px x 2 rows). Wave = (row, cout-stripe); each wave
// owns a full 64-px row. kpos fully unrolled; B-fragment loads staggered one
// fragment ahead of their WMMA so ds waits stay > 0; A prefetched ahead.
// ---------------------------------------------------------------------------
__global__ void __launch_bounds__(512)
conv_kernel(const unsigned short* __restrict__ inbf,
            const unsigned short* __restrict__ wm,
            float* __restrict__ out) {
    __shared__ __align__(64) unsigned short in_s[4 * 66 * 32]; // (dy, xi, cin)
    int yp  = blockIdx.x;         // row pair: rows 2yp, 2yp+1
    int mbb = blockIdx.y;         // cout block (128)
    int b   = blockIdx.z;
    int tid = threadIdx.x;
    int lane = tid & 31, wave = tid >> 5;   // 16 waves
    int lane15 = lane & 15, laneHi = lane >> 4;
    int mt = wave & 7;            // cout 16-stripe (8)
    int rt = wave >> 3;           // which of the two output rows
    int m0 = mbb * 128 + mt * 16;
    int y  = yp * 2 + rt;
    int y0 = yp * 2 - 1;          // global row behind LDS halo row 0
    v8f acc[4] = {};

    // x-halo columns (xi=0,65) are pure zero padding: write once.
    for (int idx = tid; idx < 4 * 32 * 2; idx += 512) {
        int dy = idx >> 6, c = (idx >> 1) & 31, side = idx & 1;
        in_s[(dy * 66 + (side ? 65 : 0)) * 32 + c] = 0;
    }

    for (int kc = 0; kc < CIN / 32; ++kc) {
        __syncthreads();
        // Stage 4-row interior (bf16, u32 = 2 x-values per load, coalesced),
        // scattered to cin-fastest LDS layout.
        for (int idx = tid; idx < 4 * 32 * 32; idx += 512) {
            int dy = idx >> 10;                // 0..3
            int c  = (idx >> 5) & 31;
            int xp = idx & 31;                 // x pair
            int yg = y0 + dy;
            unsigned int v = 0;
            if ((unsigned)yg < 64u)
                v = *(const unsigned int*)(inbf
                      + (((size_t)b * CIN + kc * 32 + c) * HH + yg) * WW
                      + 2 * xp);
            int lbase = (dy * 66 + 2 * xp + 1) * 32 + c;
            in_s[lbase]      = (unsigned short)(v & 0xffffu);
            in_s[lbase + 32] = (unsigned short)(v >> 16);
        }
        __syncthreads();
        // A fragments for this (b, kc, cout-stripe); kpos stride = 16*32*512
        const unsigned short* wbase =
            wm + ((size_t)(b * NKBLK + kc) * 32 + (mbb * 8 + mt)) * 512
               + lane * 16;
        #pragma unroll
        for (int kpos = 0; kpos < 9; ++kpos) {
            const int kdy = kpos / 3, kdx = kpos % 3;   // compile-time
            v16bf a = *(const v16bf*)(wbase + (size_t)kpos * (16 * 32 * 512));
            if (kpos < 8)   // hide L2 latency of next A fragment
                __builtin_prefetch(wbase + (size_t)(kpos + 1) * (16 * 32 * 512),
                                   0, 3);
            const unsigned short* lb =
                &in_s[((rt + kdy) * 66 + lane15 + kdx) * 32 + laneHi * 16];
            // staggered pipeline: fragment t+1 in flight during WMMA t
            v16bf b0 = *(const v16bf*)(lb);
            v16bf b1 = *(const v16bf*)(lb + 16 * 32);
            acc[0] = wmma_bf16(a, b0, acc[0]);
            v16bf b2 = *(const v16bf*)(lb + 32 * 32);
            acc[1] = wmma_bf16(a, b1, acc[1]);
            v16bf b3 = *(const v16bf*)(lb + 48 * 32);
            acc[2] = wmma_bf16(a, b2, acc[2]);
            acc[3] = wmma_bf16(a, b3, acc[3]);
        }
    }
    // Epilogue: C/D layout -> VGPR r: lanes0-15 (M=r), lanes16-31 (M=r+8)
    for (int r = 0; r < 8; ++r) {
        int cout = m0 + r + laneHi * 8;
        size_t base = (((size_t)b * COUT + cout) * HH + y) * WW;
        #pragma unroll
        for (int t = 0; t < 4; ++t)
            out[base + t * 16 + lane15] = acc[t][r];
    }
}

// ---------------------------------------------------------------------------
// Kernel 4: per-row inverse norms of the raw weight matrix.
// ---------------------------------------------------------------------------
__global__ void norm_kernel(const float* __restrict__ weight,
                            float* __restrict__ invn) {
    __shared__ float red[256];
    int o = blockIdx.x, tid = threadIdx.x;
    const float* wrow = weight + (size_t)o * KTOT;
    float sum = 0.f;
    for (int j = tid; j < KTOT; j += 256) { float v = wrow[j]; sum += v * v; }
    red[tid] = sum; __syncthreads();
    for (int s = 128; s > 0; s >>= 1) {
        if (tid < s) red[tid] += red[tid + s];
        __syncthreads();
    }
    if (tid == 0) invn[o] = 1.f / fmaxf(sqrtf(red[0]), 1e-12f);
}

// ---------------------------------------------------------------------------
// Kernel 5: co = (W/|W|) @ (W/|W|)^T, fp32 LDS-tiled (exactness-preserving).
// ---------------------------------------------------------------------------
__global__ void __launch_bounds__(256)
co_kernel(const float* __restrict__ weight,
          const float* __restrict__ invn,
          float* __restrict__ co) {
    __shared__ float a_s[32][33];
    __shared__ float b_s[32][33];
    int i0 = blockIdx.y * 32, j0 = blockIdx.x * 32;
    int tid = threadIdx.x;
    int tx = tid & 31, ty = tid >> 5;
    float acc[4] = {0.f, 0.f, 0.f, 0.f};
    for (int k0 = 0; k0 < KTOT; k0 += 32) {
        __syncthreads();
        for (int p = 0; p < 4; ++p) {
            int e = tid + p * 256;
            int row = e >> 5, kk = e & 31;
            a_s[row][kk] = weight[(size_t)(i0 + row) * KTOT + k0 + kk];
            b_s[row][kk] = weight[(size_t)(j0 + row) * KTOT + k0 + kk];
        }
        __syncthreads();
        for (int kk = 0; kk < 32; ++kk) {
            float bv = b_s[tx][kk];
            acc[0] += a_s[ty     ][kk] * bv;
            acc[1] += a_s[ty +  8][kk] * bv;
            acc[2] += a_s[ty + 16][kk] * bv;
            acc[3] += a_s[ty + 24][kk] * bv;
        }
    }
    for (int p = 0; p < 4; ++p) {
        int i = i0 + ty + p * 8;
        co[(size_t)i * COUT + j0 + tx] = acc[p] * invn[i] * invn[j0 + tx];
    }
}

// ---------------------------------------------------------------------------
extern "C" void kernel_launch(void* const* d_in, const int* in_sizes, int n_in,
                              void* d_out, int out_size, void* d_ws, size_t ws_size,
                              hipStream_t stream) {
    (void)in_sizes; (void)n_in; (void)out_size; (void)ws_size;
    const float* input  = (const float*)d_in[0];
    const float* prob   = (const float*)d_in[1];
    const float* style  = (const float*)d_in[2];
    const float* weight = (const float*)d_in[3];   // (1,512,512,3,3)
    const float* mw     = (const float*)d_in[4];
    const float* mb     = (const float*)d_in[5];

    float* out = (float*)d_out;                              // (16,512,64,64)
    float* co  = out + (size_t)NB * COUT * HH * WW;          // (1,512,512)

    // workspace layout
    float*          s_buf = (float*)d_ws;                    // 16*512 floats
    float*          invn  = s_buf + NB * CIN;                // 512 floats
    unsigned short* wm    = (unsigned short*)((char*)d_ws + 65536);     // 75.5MB
    unsigned short* inbf  = (unsigned short*)((char*)d_ws + 65536
                                              + (size_t)NB * NKBLK * 32 * 512 * 2);

    const int n4 = NB * CIN * HH * WW / 4;                   // 8,388,608
    cvt_kernel  <<<n4 / 256, 256, 0, stream>>>(input, inbf);
    style_kernel<<<NB * CIN / 256, 256, 0, stream>>>(style, prob, mw, mb, s_buf);
    wprep_kernel<<<dim3(COUT, NB), 256, 0, stream>>>(weight, s_buf, wm);
    conv_kernel <<<dim3(HH / 2, COUT / 128, NB), 512, 0, stream>>>(inbf, wm, out);
    norm_kernel <<<COUT, 256, 0, stream>>>(weight, invn);
    co_kernel   <<<dim3(COUT / 32, COUT / 32), 256, 0, stream>>>(weight, invn, co);
}
